// SelfAttention_867583394316
// MI455X (gfx1250) — matrix operand order repaired and checked by reading, not compiled, not run on hardware
//
#include <hip/hip_runtime.h>

// Self-attention (S=2048, B=4, E=1024, H=16, d=64) on gfx1250, bf16 WMMA path.
// prep: x->bf16, Wqkv/Wout -> bf16 transposed ->
// gemm1: qkv = x@Wqkv + bqkv (bf16) -> transpose V ->
// flash attention (transposed-score & transposed-O WMMA, online softmax) ->
// gemm2: out = o@Wout + bout (f32)

typedef __attribute__((ext_vector_type(16))) __bf16 v16bf;
typedef __attribute__((ext_vector_type(8)))  __bf16 v8bf;
typedef __attribute__((ext_vector_type(4)))  __bf16 v4bf;
typedef __attribute__((ext_vector_type(8)))  float  v8f;
typedef __attribute__((ext_vector_type(4)))  float  v4f;
typedef __attribute__((ext_vector_type(4)))  int    v4i;

#define S_LEN   2048
#define B_SZ    4
#define E_DIM   1024
#define H_HEADS 16
#define D_HEAD  64
#define N3E     (3 * E_DIM)

static __device__ __forceinline__ __bf16 f2bf(float f) { return (__bf16)f; }
static __device__ __forceinline__ float  bf2f(__bf16 h) { return (float)h; }
static __device__ __forceinline__ void st_out(float* p, float v)  { *p = v; }
static __device__ __forceinline__ void st_out(__bf16* p, float v) { *p = f2bf(v); }

static __device__ __forceinline__ v8bf ld8(const __bf16* p) { return *(const v8bf*)p; }
static __device__ __forceinline__ v16bf combine(v8bf lo, v8bf hi) {
  return __builtin_shufflevector(lo, hi, 0, 1, 2, 3, 4, 5, 6, 7,
                                 8, 9, 10, 11, 12, 13, 14, 15);
}

// cross-half (xor 16) exchange on VALU via v_permlanex16 (identity selects)
static __device__ __forceinline__ float xor16f(float v) {
  int d = __builtin_bit_cast(int, v);
  d = __builtin_amdgcn_permlanex16(d, d, 0x76543210, 0xfedcba98, false, false);
  return __builtin_bit_cast(float, d);
}
static __device__ __forceinline__ v8bf xor16v(v8bf v) {
  v4i d = __builtin_bit_cast(v4i, v);
#pragma unroll
  for (int i = 0; i < 4; ++i)
    d[i] = __builtin_amdgcn_permlanex16(d[i], d[i], 0x76543210, 0xfedcba98,
                                        false, false);
  return __builtin_bit_cast(v8bf, d);
}

// ---------------------------------------------------------------------------
// Elementwise f32 -> bf16 (vectorized)
// ---------------------------------------------------------------------------
__global__ __launch_bounds__(256)
void convert_bf16(const float* __restrict__ in, __bf16* __restrict__ out)
{
  const size_t i = ((size_t)blockIdx.x * blockDim.x + threadIdx.x) * 4;
  v4f v = *(const v4f*)(in + i);
  v4bf r;
#pragma unroll
  for (int c = 0; c < 4; ++c) r[c] = f2bf(v[c]);
  *(v4bf*)(out + i) = r;
}

// ---------------------------------------------------------------------------
// Weight transpose: in[K][N] f32 -> out[N][K] bf16 (LDS tiled)
// ---------------------------------------------------------------------------
__global__ __launch_bounds__(256)
void transpose_w(const float* __restrict__ in, __bf16* __restrict__ out,
                 int K, int N)
{
  __shared__ __bf16 t[64][66];
  const int nblk = N >> 6;
  const int n0 = (blockIdx.x % nblk) << 6;
  const int k0 = (blockIdx.x / nblk) << 6;
  const int tx = threadIdx.x & 63;
  const int ty = threadIdx.x >> 6;
#pragma unroll
  for (int i = 0; i < 16; ++i)
    t[i * 4 + ty][tx] = f2bf(in[(size_t)(k0 + i * 4 + ty) * N + n0 + tx]);
  __syncthreads();
#pragma unroll
  for (int i = 0; i < 16; ++i)
    out[(size_t)(n0 + i * 4 + ty) * K + k0 + tx] = t[tx][i * 4 + ty];
}

// ---------------------------------------------------------------------------
// V transpose: qkv V region (s,b,h,dd) bf16 -> vt[(b*H+h)*64+dd][s]
// ---------------------------------------------------------------------------
__global__ __launch_bounds__(256)
void transpose_v(const __bf16* __restrict__ qkv, __bf16* __restrict__ vt)
{
  __shared__ __bf16 t[64][66];
  const int bh = blockIdx.x >> 5;
  const int s0 = (blockIdx.x & 31) * 64;
  const int b = bh >> 4, h = bh & 15;
  const int tx = threadIdx.x & 63;
  const int ty = threadIdx.x >> 6;
#pragma unroll
  for (int i = 0; i < 16; ++i) {
    const int sl = i * 4 + ty;
    t[sl][tx] = qkv[((size_t)(s0 + sl) * B_SZ + b) * N3E + 2 * E_DIM
                    + h * D_HEAD + tx];
  }
  __syncthreads();
#pragma unroll
  for (int i = 0; i < 16; ++i) {
    const int dd = i * 4 + ty;
    vt[((size_t)(bh * D_HEAD + dd)) * S_LEN + s0 + tx] = t[tx][dd];
  }
}

// ---------------------------------------------------------------------------
// GEMM: C(MxN) = A(MxK) @ Bt^T + bias, A bf16 row-major, Bt = B transposed
// [N][K] bf16. One wave = 16x64 strip, 4 WMMA accumulators, all operand
// loads contiguous 16/32B vectors.
// ---------------------------------------------------------------------------
template <typename OT>
__global__ __launch_bounds__(256)
void gemm_bias_wmma(const __bf16* __restrict__ A, const __bf16* __restrict__ Bt,
                    const float* __restrict__ bias, OT* __restrict__ C,
                    int M, int N, int K)
{
  const int wslot = threadIdx.x >> 5;
  const int lane  = threadIdx.x & 31;
  const int half  = lane >> 4;
  const int l16   = lane & 15;
  const int mblocks = M >> 7;                      // (M/16)/8
  const int tm = ((blockIdx.x % mblocks) * 8 + wslot) << 4;
  const int tn = (blockIdx.x / mblocks) << 6;

  const __bf16* arow = A + (size_t)(tm + l16) * K;     // A: lane = row
  const __bf16* bcol[4];
#pragma unroll
  for (int nt = 0; nt < 4; ++nt)                       // Bt: lane = column
    bcol[nt] = Bt + (size_t)(tn + nt * 16 + l16) * K + half * 16;

  v8f acc[4] = {};
#pragma unroll 2
  for (int k0 = 0; k0 < K; k0 += 32) {
    const __bf16* ap = arow + k0 + half * 16;          // A 16x32 fragment
    v16bf a = combine(ld8(ap), ld8(ap + 8));
#pragma unroll
    for (int nt = 0; nt < 4; ++nt) {
      v16bf b = *(const v16bf*)(bcol[nt] + k0);        // elem e -> k=half*16+e
      acc[nt] = __builtin_amdgcn_wmma_f32_16x16x32_bf16(false, a, false, b,
                                                        (short)0, acc[nt],
                                                        false, false);
    }
  }
#pragma unroll
  for (int nt = 0; nt < 4; ++nt) {
    const int n = tn + nt * 16 + l16;
    const float bv = bias[n];
#pragma unroll
    for (int r = 0; r < 8; ++r)                        // D: VGPR r -> row half*8+r
      st_out(C + (size_t)(tm + half * 8 + r) * N + n, acc[nt][r] + bv);
  }
}

// ---------------------------------------------------------------------------
// Flash attention, one wave per (batch*head, 16-query block).
// Scores transposed:  S^T = K_tile(16x64) @ Q^T(64x16)  -> lane = query.
// Output transposed:  O^T += V^T_tile(16x32) @ P^T(32x16) -> lane = query.
// So softmax stats, alpha-rescale and 1/l are pure per-lane scalars; the only
// cross-lane traffic is 2 scalar half-swaps + a 4-dword half-swap for P^T,
// all on v_permlanex16 (VALU).
// ---------------------------------------------------------------------------
__global__ __launch_bounds__(256)
void flash_attn_wmma(const __bf16* __restrict__ qkv,
                     const __bf16* __restrict__ vt,
                     __bf16* __restrict__ obuf)
{
  const int wave  = blockIdx.x * 8 + (threadIdx.x >> 5);
  const int qblocks = S_LEN / 16;
  const int bh = wave / qblocks;
  const int qb = wave % qblocks;
  const int b  = bh / H_HEADS;
  const int h  = bh % H_HEADS;
  const int s0 = qb * 16;
  const int lane = threadIdx.x & 31;
  const int half = lane >> 4;
  const int l16  = lane & 15;

  const float SCL2 = 0.125f * 1.44269504088896340736f;  // 1/sqrt(d) * log2(e)

  // Q^T B-fragments (lane = query column l16), pre-scaled, resident all sweep
  const __bf16* qrow = qkv + ((size_t)(s0 + l16) * B_SZ + b) * N3E + h * D_HEAD;
  v16bf qbf[2];
#pragma unroll
  for (int kk = 0; kk < 2; ++kk) {
    v16bf qraw = *(const v16bf*)(qrow + kk * 32 + half * 16);
#pragma unroll
    for (int e = 0; e < 16; ++e) qbf[kk][e] = f2bf(bf2f(qraw[e]) * SCL2);
  }

  const __bf16* vtb = vt + (size_t)bh * D_HEAD * S_LEN;

  v8f o[4] = {};                 // O^T: VGPR r -> d = nt*16+half*8+r, lane = query
  float rowm = -1e30f, rowl = 0.f;   // per-lane stats of query l16

#pragma unroll 2
  for (int j0 = 0; j0 < S_LEN; j0 += 32) {
    if (j0 + 32 < S_LEN) {                // prefetch next K rows + V^T rows
      __builtin_prefetch(qkv + ((size_t)(j0 + 32 + lane) * B_SZ + b) * N3E
                         + E_DIM + h * D_HEAD, 0, 0);
      __builtin_prefetch(vtb + (size_t)lane * S_LEN + j0 + 32, 0, 0);
      __builtin_prefetch(vtb + (size_t)(lane + 32) * S_LEN + j0 + 32, 0, 0);
    }
    // ---- issue all fragment loads up front ----
    v16bf ka[4];                          // K A-fragments: lane = key row
#pragma unroll
    for (int t = 0; t < 2; ++t) {
      const __bf16* kp = qkv + ((size_t)(j0 + t * 16 + l16) * B_SZ + b) * N3E
                       + E_DIM + h * D_HEAD;
#pragma unroll
      for (int kk = 0; kk < 2; ++kk) {
        const __bf16* p = kp + kk * 32 + half * 16;
        ka[t * 2 + kk] = combine(ld8(p), ld8(p + 8));
      }
    }
    v16bf va[4];                          // V^T A-fragments: lane = d row
#pragma unroll
    for (int nt = 0; nt < 4; ++nt) {
      const __bf16* vp = vtb + (size_t)(nt * 16 + l16) * S_LEN + j0 + half * 16;
      va[nt] = combine(ld8(vp), ld8(vp + 8));
    }
    // ---- transposed scores: lane = query, VGPR r = key j0 + t*16 + half*8+r
    v8f sct[2];
#pragma unroll
    for (int t = 0; t < 2; ++t) {
      v8f s = {};
      s = __builtin_amdgcn_wmma_f32_16x16x32_bf16(false, ka[t * 2 + 0], false,
                                                  qbf[0], (short)0, s, false, false);
      s = __builtin_amdgcn_wmma_f32_16x16x32_bf16(false, ka[t * 2 + 1], false,
                                                  qbf[1], (short)0, s, false, false);
      sct[t] = s;
    }
    // ---- online softmax: register-local reduce + one VALU half-swap ----
    float mloc = fmaxf(sct[0][0], sct[1][0]);
#pragma unroll
    for (int e = 1; e < 8; ++e) mloc = fmaxf(mloc, fmaxf(sct[0][e], sct[1][e]));
    float mx   = fmaxf(mloc, xor16f(mloc));
    float newm = fmaxf(rowm, mx);
    float alpha = __builtin_amdgcn_exp2f(rowm - newm);
    rowm = newm;
    v8bf lo8, hi8;                        // p of this lane's 16 keys (bf16)
    float psum = 0.f;
#pragma unroll
    for (int e = 0; e < 8; ++e) {
      float p0 = __builtin_amdgcn_exp2f(sct[0][e] - newm);
      float p1 = __builtin_amdgcn_exp2f(sct[1][e] - newm);
      psum += p0 + p1;
      lo8[e] = f2bf(p0); hi8[e] = f2bf(p1);
    }
    psum += xor16f(psum);
    rowl = rowl * alpha + psum;
    // ---- build P^T B-fragment: one 4-dword cross-half exchange ----
    // half0 owns keys {0-7,16-23}, half1 owns {8-15,24-31}; B-frag wants
    // keys half*16+e, so half0 sends hi8 (16-23), half1 sends lo8 (8-15).
    v8bf send = half ? lo8 : hi8;
    v8bf recv = xor16v(send);
    v16bf pb  = half ? combine(recv, hi8) : combine(lo8, recv);
    // ---- rescale O^T by per-lane alpha, then O^T += V^T @ P^T ----
#pragma unroll
    for (int nt = 0; nt < 4; ++nt) {
#pragma unroll
      for (int r = 0; r < 8; ++r) o[nt][r] *= alpha;
      o[nt] = __builtin_amdgcn_wmma_f32_16x16x32_bf16(false, va[nt], false, pb,
                                                      (short)0, o[nt],
                                                      false, false);
    }
  }
  // ---- normalize (per-lane) and store: 4 x 16B vector stores per lane ----
  const float inv = 1.0f / rowl;
  __bf16* orow = obuf + ((size_t)(s0 + l16) * B_SZ + b) * E_DIM + h * D_HEAD;
#pragma unroll
  for (int nt = 0; nt < 4; ++nt) {
    v8bf ov;
#pragma unroll
    for (int r = 0; r < 8; ++r) ov[r] = f2bf(o[nt][r] * inv);
    *(v8bf*)(orow + nt * 16 + half * 8) = ov;
  }
}

// ---------------------------------------------------------------------------
extern "C" void kernel_launch(void* const* d_in, const int* in_sizes, int n_in,
                              void* d_out, int out_size, void* d_ws, size_t ws_size,
                              hipStream_t stream)
{
  (void)in_sizes; (void)n_in; (void)out_size; (void)ws_size;
  const float* x    = (const float*)d_in[0];
  const float* Wqkv = (const float*)d_in[1];
  const float* bqkv = (const float*)d_in[2];
  const float* Wout = (const float*)d_in[3];
  const float* bout = (const float*)d_in[4];
  float* out = (float*)d_out;

  const int M = S_LEN * B_SZ;                              // 8192 rows
  __bf16* qkv   = (__bf16*)d_ws;                           // M x 3E      (48 MB)
  __bf16* obuf  = qkv   + (size_t)M * N3E;                 // M x E       (16 MB)
  __bf16* vt    = obuf  + (size_t)M * E_DIM;               // 64*64 x S   (16 MB)
  __bf16* xbf   = vt    + (size_t)B_SZ * E_DIM * S_LEN;    // M x E       (16 MB)
  __bf16* wqkvt = xbf   + (size_t)M * E_DIM;               // 3E x E      ( 6 MB)
  __bf16* woutt = wqkvt + (size_t)N3E * E_DIM;             // E x E       ( 2 MB)

  // prep: x -> bf16; weights -> bf16 transposed [N][K]
  convert_bf16<<<(int)(((size_t)M * E_DIM) / 1024), 256, 0, stream>>>(x, xbf);
  transpose_w<<<(N3E >> 6) * (E_DIM >> 6), 256, 0, stream>>>(Wqkv, wqkvt, E_DIM, N3E);
  transpose_w<<<(E_DIM >> 6) * (E_DIM >> 6), 256, 0, stream>>>(Wout, woutt, E_DIM, E_DIM);

  // QKV projection
  gemm_bias_wmma<__bf16><<<(M >> 7) * (N3E >> 6), 256, 0, stream>>>(
      xbf, wqkvt, bqkv, qkv, M, N3E, E_DIM);
  // V transpose
  transpose_v<<<64 * 32, 256, 0, stream>>>(qkv, vt);
  // Flash attention: 64 heads x 128 query blocks = 8192 waves
  flash_attn_wmma<<<(B_SZ * H_HEADS) * (S_LEN / 16) / 8, 256, 0, stream>>>(
      qkv, vt, obuf);
  // Output projection
  gemm_bias_wmma<float><<<(M >> 7) * (E_DIM >> 6), 256, 0, stream>>>(
      obuf, woutt, bout, out, M, E_DIM, E_DIM);
}
